// SparseDict_14027363188796
// MI455X (gfx1250) — matrix operand order repaired and checked by reading, not compiled, not run on hardware
//
#include <hip/hip_runtime.h>
#include <hip/hip_bf16.h>

// ---------------------------------------------------------------------------
// ISTA sparse coding, fused for MI455X (gfx1250, wave32, WMMA).
//   grad = (Z Wd^T - X) Wd  ==  Z (Wd^T Wd) - X Wd
// so precompute C = Wd^T Wd (symmetric, 512x512, bf16, lives in L2) and
// V = X Wd once; each of the 50 iterations is a single [Bt,512]x[512,512]
// GEMM done with v_wmma_f32_16x16x32_bf16 out of LDS (Z ping-pong) + L2 (C).
// ---------------------------------------------------------------------------

typedef __attribute__((ext_vector_type(16))) __bf16 v16bf;
typedef __attribute__((ext_vector_type(8)))  __bf16 bf16x8;
typedef __attribute__((ext_vector_type(8)))  float  v8f;

#define NB    8192      // batch
#define ND    1024      // D = H*W*C
#define NK    512       // K (codes)
#define BT    32        // batch rows per workgroup
#define STR   (NK + 8)  // padded LDS row stride (halves) to break bank conflicts

__device__ __forceinline__ v8f zero8() {
  v8f z;
#pragma unroll
  for (int i = 0; i < 8; ++i) z[i] = 0.0f;
  return z;
}

// A fragment (16x32 bf16): lane lr holds row M=lr; K segment
// [k0 + hf*16, +16) contiguous -> two b128 loads.
__device__ __forceinline__ v16bf load_a_frag(const __bf16* rowptr, int k0, int hf) {
  bf16x8 lo = *(const bf16x8*)(rowptr + k0 + hf * 16);
  bf16x8 hi = *(const bf16x8*)(rowptr + k0 + hf * 16 + 8);
  return __builtin_shufflevector(lo, hi, 0,1,2,3,4,5,6,7,8,9,10,11,12,13,14,15);
}

// B fragment (32x16 bf16): lane lr holds column N=lr; K segment
// [k0 + hf*16, +16) contiguous -> 32B load (2x b128).
__device__ __forceinline__ v16bf load_b_frag(const __bf16* rowptr, int k0, int hf) {
  return *(const v16bf*)(rowptr + k0 + hf * 16);
}

__device__ __forceinline__ v8f wmma_bf16(v16bf a, v16bf b, v8f c) {
  return __builtin_amdgcn_wmma_f32_16x16x32_bf16(false, a, false, b, (short)0, c,
                                                 false, false);
}

// --- prep: dictionary f32 -> bf16 row-major + transposed copy --------------
__global__ void prep_dict(const float* __restrict__ Wd, __bf16* __restrict__ Wdb,
                          __bf16* __restrict__ WdTb) {
  int idx = blockIdx.x * blockDim.x + threadIdx.x;     // over 1024*512
  if (idx >= ND * NK) return;
  int d = idx >> 9, k = idx & (NK - 1);
  float w = Wd[idx];
  Wdb[idx] = (__bf16)w;
  WdTb[(size_t)k * ND + d] = (__bf16)w;
}

// --- prep: z_e_x [B,C,H,W] -> X_bf [B, H*W*C] (permute 0,2,3,1), bf16 ------
__global__ void prep_x(const float* __restrict__ z, __bf16* __restrict__ Xb) {
  size_t idx = (size_t)blockIdx.x * blockDim.x + threadIdx.x;   // over 8192*1024
  if (idx >= (size_t)NB * ND) return;
  int b = (int)(idx >> 10), dd = (int)(idx & (ND - 1));
  int c = dd & 15, wq = (dd >> 4) & 7, h = dd >> 7;
  Xb[idx] = (__bf16)z[(size_t)b * ND + c * 64 + h * 8 + wq];
}

// --- Gram: C = Wd^T Wd via WMMA; one 16x16 tile per wave -------------------
__global__ __launch_bounds__(256)
void gram_kernel(const __bf16* __restrict__ WdTb, __bf16* __restrict__ Cb) {
  const int wv = threadIdx.x >> 5, lane = threadIdx.x & 31;
  const int lr = lane & 15, hf = lane >> 4;
  const int t = blockIdx.x * 8 + wv;          // 0..1023 tiles (32x32)
  const int i0 = (t >> 5) * 16, j0 = (t & 31) * 16;
  const __bf16* arow = WdTb + (size_t)(i0 + lr) * ND;   // row i of Wd^T
  const __bf16* brow = WdTb + (size_t)(j0 + lr) * ND;   // col j of Wd = row j of Wd^T
  v8f acc = zero8();
  for (int k0 = 0; k0 < ND; k0 += 32)
    acc = wmma_bf16(load_a_frag(arow, k0, hf), load_b_frag(brow, k0, hf), acc);
#pragma unroll
  for (int r = 0; r < 8; ++r)
    Cb[(size_t)(i0 + r + 8 * hf) * NK + j0 + lr] = (__bf16)acc[r];
}

// --- Power iteration on C -> step = 1/L, thr = alpha/L; zero sparsity slot -
__global__ __launch_bounds__(512)
void power_kernel(const __bf16* __restrict__ Cb, float* __restrict__ scal,
                  float* __restrict__ slot) {
  __shared__ float v[NK], w[NK];
  __shared__ float snorm;
  const int i = threadIdx.x;
  v[i] = 1.0f;
  if (i == 0) *slot = 0.0f;                    // sparsity accumulator (graph-safe)
  __syncthreads();
  const __bf16* row = Cb + (size_t)i * NK;
  for (int it = 0; it < 20; ++it) {
    float s = 0.0f;
    for (int j = 0; j < NK; ++j) s += (float)row[j] * v[j];
    w[i] = s;
    __syncthreads();
    if (i == 0) {
      float n = 0.0f;
      for (int j = 0; j < NK; ++j) n += w[j] * w[j];
      snorm = __builtin_sqrtf(n) + 1e-12f;
    }
    __syncthreads();
    v[i] = w[i] / snorm;
    __syncthreads();
  }
  float s = 0.0f;
  for (int j = 0; j < NK; ++j) s += (float)row[j] * v[j];
  w[i] = s;
  __syncthreads();
  if (i == 0) {
    float num = 0.0f, den = 0.0f;
    for (int j = 0; j < NK; ++j) { num += v[j] * w[j]; den += v[j] * v[j]; }
    float L = num / (den + 1e-12f);
    scal[0] = 1.0f / L;          // step
    scal[1] = 0.1f / L;          // step * ALPHA
  }
}

// --- Fused: V = X Wd, 50 ISTA iterations, sparsity, reconstruction --------
__global__ __launch_bounds__(256)
void ista_fused(const __bf16* __restrict__ Xb, const __bf16* __restrict__ WdTb,
                const __bf16* __restrict__ Wdb, const __bf16* __restrict__ Cb,
                const float* __restrict__ scal, float* __restrict__ out,
                float* __restrict__ slot) {
  __shared__ __bf16 Zb[2][BT * STR];  // ping-pong Z (Jacobi update)
  __shared__ __bf16 Vb[BT * STR];     // V = X Wd for this row block
  __shared__ int s_cnt;

  const int tid = threadIdx.x;
  const int wv = tid >> 5, lane = tid & 31, lr = lane & 15, hf = lane >> 4;
  const int mg = wv >> 2, ng = wv & 3;             // 2 M-groups x 4 N-groups
  const int bBase = blockIdx.x * BT;
  const float step = scal[0], thr = scal[1];

  for (int i = tid; i < BT * STR; i += 256) Zb[0][i] = (__bf16)0.0f;
  if (tid == 0) s_cnt = 0;

  // Phase 0: V[rows, ng*128 .. +128] = X Wd  (K-dim = 1024)
  {
    v8f acc[8];
#pragma unroll
    for (int t = 0; t < 8; ++t) acc[t] = zero8();
    const __bf16* arow = Xb + (size_t)(bBase + mg * 16 + lr) * ND;
    for (int k0 = 0; k0 < ND; k0 += 32) {
      v16bf a = load_a_frag(arow, k0, hf);
#pragma unroll
      for (int t = 0; t < 8; ++t) {
        const __bf16* brow = WdTb + (size_t)(ng * 128 + t * 16 + lr) * ND;
        acc[t] = wmma_bf16(a, load_b_frag(brow, k0, hf), acc[t]);
      }
    }
#pragma unroll
    for (int t = 0; t < 8; ++t)
#pragma unroll
      for (int r = 0; r < 8; ++r)
        Vb[(mg * 16 + r + 8 * hf) * STR + ng * 128 + t * 16 + lr] = (__bf16)acc[t][r];
  }
  __syncthreads();

  // Phase 1: 50 ISTA iterations. G = Z C (C symmetric: column n == row n).
  for (int it = 0; it < 50; ++it) {
    __bf16* cur = Zb[it & 1];
    __bf16* nxt = Zb[(it + 1) & 1];
    v8f acc[8];
#pragma unroll
    for (int t = 0; t < 8; ++t) acc[t] = zero8();
    const __bf16* arow = cur + (mg * 16 + lr) * STR;
    for (int k0 = 0; k0 < NK; k0 += 32) {
      v16bf a = load_a_frag(arow, k0, hf);
#pragma unroll
      for (int t = 0; t < 8; ++t) {
        const __bf16* brow = Cb + (size_t)(ng * 128 + t * 16 + lr) * NK;
        acc[t] = wmma_bf16(a, load_b_frag(brow, k0, hf), acc[t]);
      }
    }
#pragma unroll
    for (int t = 0; t < 8; ++t)
#pragma unroll
      for (int r = 0; r < 8; ++r) {
        const int idx = (mg * 16 + r + 8 * hf) * STR + ng * 128 + t * 16 + lr;
        float z = (float)cur[idx];
        float v = (float)Vb[idx];
        float zn = z - step * (acc[t][r] - v);
        float mag = fabsf(zn) - thr;
        nxt[idx] = (__bf16)(mag > 0.0f ? copysignf(mag, zn) : 0.0f);
      }
    __syncthreads();
  }
  // final Z is in Zb[0] (50 is even)

  // Phase 2: sparsity count
  int local = 0;
  for (int i = tid; i < BT * NK; i += 256) {
    int row = i >> 9, col = i & (NK - 1);
    if (fabsf((float)Zb[0][row * STR + col]) > 1e-6f) ++local;
  }
  atomicAdd(&s_cnt, local);
  __syncthreads();
  if (tid == 0) atomicAdd(slot, (float)s_cnt * (1.0f / (float)NB));

  // Phase 3: reconstruction out = Z Wd^T  (col d of Wd^T = row d of Wd)
  {
    const __bf16* arow = Zb[0] + (mg * 16 + lr) * STR;
#pragma unroll
    for (int pass = 0; pass < 2; ++pass) {
      const int cbase = ng * 256 + pass * 128;
      v8f acc[8];
#pragma unroll
      for (int t = 0; t < 8; ++t) acc[t] = zero8();
      for (int k0 = 0; k0 < NK; k0 += 32) {
        v16bf a = load_a_frag(arow, k0, hf);
#pragma unroll
        for (int t = 0; t < 8; ++t) {
          const __bf16* brow = Wdb + (size_t)(cbase + t * 16 + lr) * NK;
          acc[t] = wmma_bf16(a, load_b_frag(brow, k0, hf), acc[t]);
        }
      }
#pragma unroll
      for (int t = 0; t < 8; ++t)
#pragma unroll
        for (int r = 0; r < 8; ++r)
          out[(size_t)(bBase + mg * 16 + r + 8 * hf) * ND + cbase + t * 16 + lr] =
              acc[t][r];
    }
  }
}

extern "C" void kernel_launch(void* const* d_in, const int* in_sizes, int n_in,
                              void* d_out, int out_size, void* d_ws, size_t ws_size,
                              hipStream_t stream) {
  const float* z    = (const float*)d_in[0];   // [8192,16,8,8] f32
  const float* dict = (const float*)d_in[1];   // [1024,512]   f32
  float* out = (float*)d_out;                  // [8192*1024] + 1 (sparsity)

  char* ws = (char*)d_ws;
  size_t off = 0;
  __bf16* Xb   = (__bf16*)(ws + off); off += (size_t)NB * ND * 2;   // 16 MB
  __bf16* Wdb  = (__bf16*)(ws + off); off += (size_t)ND * NK * 2;   //  1 MB
  __bf16* WdTb = (__bf16*)(ws + off); off += (size_t)NK * ND * 2;   //  1 MB
  __bf16* Cb   = (__bf16*)(ws + off); off += (size_t)NK * NK * 2;   // 512 KB
  float*  scal = (float*)(ws + off);                                // step, thr

  float* slot = out + (size_t)NB * ND;         // sparsity scalar

  prep_dict<<<(ND * NK + 255) / 256, 256, 0, stream>>>(dict, Wdb, WdTb);
  prep_x<<<(int)(((size_t)NB * ND + 255) / 256), 256, 0, stream>>>(z, Xb);
  gram_kernel<<<(NK / 16) * (NK / 16) / 8, 256, 0, stream>>>(WdTb, Cb);
  power_kernel<<<1, 512, 0, stream>>>(Cb, scal, slot);
  ista_fused<<<NB / BT, 256, 0, stream>>>(Xb, WdTb, Wdb, Cb, scal, out, slot);
}
